// SelfAttentionHead_50062138802611
// MI455X (gfx1250) — matrix operand order repaired and checked by reading, not compiled, not run on hardware
//
#include <hip/hip_runtime.h>

typedef __bf16 bf16_t;
typedef __attribute__((ext_vector_type(16))) __bf16 v16bf;
typedef __attribute__((ext_vector_type(8)))  __bf16 v8bf;
typedef __attribute__((ext_vector_type(8)))  float   v8f;
typedef __attribute__((ext_vector_type(4)))  float   v4f;
typedef __attribute__((ext_vector_type(4)))  unsigned int v4u;
typedef __attribute__((ext_vector_type(8)))  int     v8i;
typedef __attribute__((ext_vector_type(4)))  int     v4i;

static constexpr int Bn = 8, Tn = 2048, Cn = 1024, Hn = 64;

static __device__ __forceinline__ v16bf cat8(v8bf a, v8bf b) {
  v16bf r;
#pragma unroll
  for (int i = 0; i < 8; ++i) { r[i] = a[i]; r[i + 8] = b[i]; }
  return r;
}

// A fragment (16x32 bf16, M x K): lane&15 = row M.
// lanes 0-15 supply K = {0..7, 16..23}; lanes 16-31 supply K = {8..15, 24..31}.
static __device__ __forceinline__ v16bf load_fragA(const bf16_t* rowp, int hi) {
  int o = hi ? 8 : 0;
  return cat8(*(const v8bf*)(rowp + o), *(const v8bf*)(rowp + o + 16));
}

// B fragment (32x16 bf16, K x N): lane&15 = column N.
// lanes 0-15 supply K = {0..7, 16..23}; lanes 16-31 supply K = {8..15, 24..31}.
static __device__ __forceinline__ v16bf load_fragB(const bf16_t* colp, int hi) {
  int o = hi ? 8 : 0;
  return cat8(*(const v8bf*)(colp + o), *(const v8bf*)(colp + o + 16));
}

static __device__ __forceinline__ v8f wmma_bf16(v16bf a, v16bf b, v8f c) {
  return __builtin_amdgcn_wmma_f32_16x16x32_bf16(false, a, false, b,
                                                 (short)0, c, false, false);
}

// gfx1250 async copy: 16 bytes global -> LDS, tracked by ASYNCcnt.
static __device__ __forceinline__ void async_copy16(const bf16_t* g, bf16_t* l) {
  unsigned lds_addr = (unsigned)(uintptr_t)l;          // low 32 bits = LDS offset
  unsigned long long ga = (unsigned long long)(uintptr_t)g;
  asm volatile("global_load_async_to_lds_b128 %0, %1, off"
               :: "v"(lds_addr), "v"(ga) : "memory");
}
static __device__ __forceinline__ void wait_asynccnt0() {
  asm volatile("s_wait_asynccnt 0" ::: "memory");
}

// TDM: load a 2D tile (rows x row_elems bf16, row stride = tensor row stride)
// from global to LDS, hardware-padding each row so the LDS stride is
// row_elems + 8 bf16. pad_interval_enc: row bytes = 8 * 2^pad_interval.
static __device__ __forceinline__ void tdm_load_tile(const bf16_t* gsrc,
                                                     bf16_t* ldst,
                                                     int rows, int row_elems,
                                                     int row_stride_elems,
                                                     int pad_interval_enc) {
  unsigned lds_addr = (unsigned)(uintptr_t)ldst;
  unsigned long long ga = (unsigned long long)(uintptr_t)gsrc;
  v4u g0 = { 1u,                                   // count=1, user descriptor
             lds_addr,                             // lds_addr
             (unsigned)(ga & 0xFFFFFFFFu),         // global_addr[31:0]
             (unsigned)((ga >> 32) & 0x01FFFFFFu) | (2u << 30) };  // type=2
  v8i g1;
  g1[0] = (1 << 16)                 // data_size = 2 bytes
        | (1 << 20)                 // pad_enable
        | (pad_interval_enc << 22)  // pad every 2^(n+1) DWORDs (= one row)
        | (3 << 25);                // pad_amount = 4 DWORDs (16 B)
  g1[1] = (row_elems & 0xFFFF) << 16;  // tensor_dim0[15:0] in bits 63:48
  g1[2] = (rows & 0xFFFF) << 16;       // tensor_dim1[15:0] in bits 95:80
  g1[3] = (row_elems & 0xFFFF) << 16;  // tile_dim0 in bits 127:112
  g1[4] = rows;                        // tile_dim1 in bits 143:128
  g1[5] = row_stride_elems;            // tensor_dim0_stride[31:0]
  g1[6] = 0;
  g1[7] = 0;
  v4i gz4 = { 0, 0, 0, 0 };
  v8i gz8 = { 0, 0, 0, 0, 0, 0, 0, 0 };
  // amdgpu-toolchain (clang-23) 6-arg form:
  // (v4u g0, v8i g1, v4i g2, v4i g3, v8i, i32 cpol)
  __builtin_amdgcn_tensor_load_to_lds(g0, g1, gz4, gz4, gz8, 0);
}

// ---------------------------------------------------------------------------
// Kernel 0: transpose+convert Wk/Wq/Wv (f32 [C,H]) -> Wt bf16 [3][H][C]
// ---------------------------------------------------------------------------
__global__ __launch_bounds__(256) void wconv_kernel(const float* __restrict__ Wk,
                                                    const float* __restrict__ Wq,
                                                    const float* __restrict__ Wv,
                                                    bf16_t* __restrict__ Wt) {
  int idx = blockIdx.x * 256 + threadIdx.x;        // 0 .. 3*64K-1
  int w = idx >> 16, r = idx & 65535;              // r = c*64 + h
  int c = r >> 6, h = r & 63;
  const float* src = (w == 0) ? Wk : (w == 1) ? Wq : Wv;
  Wt[(size_t)w * 65536 + (size_t)h * (size_t)Cn + c] = (bf16_t)src[r];
}

// ---------------------------------------------------------------------------
// Kernel 1: projections. Y[16384,192] = x[16384,1024] @ [Wk|Wq|Wv] in bf16 WMMA.
// Wave = one 16-row tile x 6 col tiles. V written transposed to Vt [B,H,T].
// ---------------------------------------------------------------------------
__global__ __launch_bounds__(256) void proj_kernel(const float* __restrict__ x,
                                                   const bf16_t* __restrict__ Wt,
                                                   bf16_t* __restrict__ Ko,
                                                   bf16_t* __restrict__ Qo,
                                                   bf16_t* __restrict__ Vto) {
  const int lane = threadIdx.x & 31;
  const int wave = threadIdx.x >> 5;
  const int hi = lane >> 4;
  const int nl = lane & 15;
  const int wid = blockIdx.x * 8 + wave;           // 0..2047
  const int rt  = wid >> 1;                        // row tile 0..1023
  const int cg  = (wid & 1) * 6;                   // col-tile group base (0 or 6)
  const int rr0 = rt * 16;

  const float* xrow = x + (size_t)(rr0 + nl) * Cn; // A row for this lane

  v8f zero = {};
  v8f acc[6];
#pragma unroll
  for (int j = 0; j < 6; ++j) acc[j] = zero;

  for (int c0 = 0; c0 < Cn; c0 += 32) {
    // A fragment: 16 f32 -> bf16 per lane
    int o = c0 + (hi ? 8 : 0);
    v4f f0 = *(const v4f*)(xrow + o);
    v4f f1 = *(const v4f*)(xrow + o + 4);
    v4f f2 = *(const v4f*)(xrow + o + 16);
    v4f f3 = *(const v4f*)(xrow + o + 20);
    v16bf a;
#pragma unroll
    for (int i = 0; i < 4; ++i) {
      a[i]      = (bf16_t)f0[i];
      a[4 + i]  = (bf16_t)f1[i];
      a[8 + i]  = (bf16_t)f2[i];
      a[12 + i] = (bf16_t)f3[i];
    }
#pragma unroll
    for (int j = 0; j < 6; ++j) {
      int ct = cg + j;                             // global col tile 0..11
      const bf16_t* colp = Wt + (size_t)(ct * 16 + nl) * Cn + c0;
      acc[j] = wmma_bf16(a, load_fragB(colp, hi), acc[j]);
    }
  }

#pragma unroll
  for (int j = 0; j < 6; ++j) {
    int ct = cg + j;
    int w  = ct >> 2;                              // 0=K, 1=Q, 2=V
    int hh = (ct & 3) * 16 + nl;
#pragma unroll
    for (int r = 0; r < 8; ++r) {
      int rr = rr0 + r + 8 * hi;                   // C-layout row
      bf16_t val = (bf16_t)acc[j][r];
      if (w == 0)      Ko[(size_t)rr * Hn + hh] = val;
      else if (w == 1) Qo[(size_t)rr * Hn + hh] = val;
      else {
        int bb = rr >> 11, t = rr & 2047;          // T = 2048
        Vto[((size_t)bb * Hn + hh) * Tn + t] = val;
      }
    }
  }
}

// ---------------------------------------------------------------------------
// Kernel 2: causal flash attention.
// Workgroup = 4 waves = 64 query rows; 32 keys per step.
// K tile (32x64) is staged by the Tensor Data Mover (tensor_load_to_lds,
// TENSORcnt, hardware LDS row padding); V^T tile (64x32) is staged with
// global_load_async_to_lds_b128 (ASYNCcnt). Both double-buffered, one step
// ahead of compute.
// ---------------------------------------------------------------------------
static constexpr int KT_STRIDE = 72;               // 64 + 8 bf16 pad (bank-safe)
static constexpr int VT_STRIDE = 40;               // 32 + 8 bf16 pad
static constexpr int PT_STRIDE = 40;               // 32 + 8 bf16 pad

__global__ __launch_bounds__(128) void flash_kernel(const bf16_t* __restrict__ Q,
                                                    const bf16_t* __restrict__ K,
                                                    const bf16_t* __restrict__ Vt,
                                                    float* __restrict__ out) {
  __shared__ __align__(16) bf16_t Kt[2][32 * KT_STRIDE];   // 2 x 4.5 KiB
  __shared__ __align__(16) bf16_t Vs[2][64 * VT_STRIDE];   // 2 x 5.0 KiB
  __shared__ __align__(16) bf16_t Pt[4][16 * PT_STRIDE];   // 4 x 1.25 KiB

  const int tid  = threadIdx.x;
  const int lane = tid & 31;
  const int wave = tid >> 5;
  const int hi = lane >> 4;
  const int nl = lane & 15;
  const int b    = blockIdx.x >> 5;
  const int q0wg = (blockIdx.x & 31) * 64;
  const int q0   = q0wg + wave * 16;

  const bf16_t* Qb = Q  + (size_t)b * Tn * Hn;
  const bf16_t* Kb = K  + (size_t)b * Tn * Hn;
  const bf16_t* Vb = Vt + (size_t)b * Hn * Tn;

  const bf16_t* qrow = Qb + (size_t)(q0 + nl) * Hn;
  v16bf qa0 = load_fragA(qrow, hi);                // dims 0..31
  v16bf qa1 = load_fragA(qrow + 32, hi);           // dims 32..63

  v8f O0 = {}, O1 = {}, O2 = {}, O3 = {};          // 16 x 64 f32 accumulator
  float m[8], l[8];
#pragma unroll
  for (int r = 0; r < 8; ++r) { m[r] = -1.0e30f; l[r] = 0.0f; }

  const float cs = 0.18033688011112042f;           // log2(e) / sqrt(64)
  bf16_t* pw = Pt[wave];

  const int kend   = q0 + 16;                      // this wave's causal bound
  const int kendwg = q0wg + 64;                    // workgroup's causal bound

  // K tile via TDM: one descriptor issued by wave 0.
  // Row = 64 bf16 = 128 B = 32 DWORDs -> pad_interval enc 4 (2^5), +16 B pad.
  auto stage_k = [&](int kb, int bufi) {
    if (wave == 0)
      tdm_load_tile(Kb + (size_t)kb * Hn, &Kt[bufi][0],
                    /*rows=*/32, /*row_elems=*/64, /*row_stride=*/Hn,
                    /*pad_interval_enc=*/4);
  };
  // V tile via async-to-LDS: 256 x 16 B chunks over 128 threads.
  auto stage_v = [&](int kb, int bufi) {
#pragma unroll
    for (int p = 0; p < 2; ++p) {
      int c = tid + p * 128;
      int vh = c >> 2, vc = (c & 3) * 8;           // 64 rows x 4 chunks
      async_copy16(Vb + (size_t)vh * Tn + kb + vc, &Vs[bufi][vh * VT_STRIDE + vc]);
    }
  };

  stage_k(0, 0);
  stage_v(0, 0);
  int buf = 0;

  for (int kb = 0; kb < kendwg; kb += 32) {
    wait_asynccnt0();                              // this wave's V stage done
    __builtin_amdgcn_s_wait_tensorcnt(0);          // wave 0's K stage done
    __syncthreads();                               // whole tile visible to WG

    if (kb + 32 < kendwg) {                        // prefetch next block
      stage_k(kb + 32, buf ^ 1);
      stage_v(kb + 32, buf ^ 1);
    }

    if (kb < kend) {                               // wave-uniform: EXEC stays full
      const bf16_t* ktb = Kt[buf];
      const bf16_t* vtb = Vs[buf];

      // ---- S = Q K^T for 2 key tiles of 16 (K-dim 64 = 2 x k32) ----
      v16bf kA0 = load_fragB(ktb + nl * KT_STRIDE, hi);
      v16bf kA1 = load_fragB(ktb + nl * KT_STRIDE + 32, hi);
      v16bf kB0 = load_fragB(ktb + (16 + nl) * KT_STRIDE, hi);
      v16bf kB1 = load_fragB(ktb + (16 + nl) * KT_STRIDE + 32, hi);

      v8f S0 = {}, S1 = {};
      S0 = wmma_bf16(qa0, kA0, S0);
      S0 = wmma_bf16(qa1, kA1, S0);
      S1 = wmma_bf16(qa0, kB0, S1);
      S1 = wmma_bf16(qa1, kB1, S1);

#pragma unroll
      for (int r = 0; r < 8; ++r) { S0[r] *= cs; S1[r] *= cs; }

      // ---- causal mask (C layout: row = q0 + r + 8*hi, col = kb (+16) + nl) ----
      if (kb + 31 > q0) {
#pragma unroll
        for (int r = 0; r < 8; ++r) {
          int row = q0 + r + 8 * hi;
          if (kb + nl > row)      S0[r] = -1.0e30f;
          if (kb + 16 + nl > row) S1[r] = -1.0e30f;
        }
      }

      // ---- online softmax: row max over 32 cols (16-lane xor reduce) ----
      float alpha[8];
#pragma unroll
      for (int r = 0; r < 8; ++r) {
        float v = fmaxf(S0[r], S1[r]);
        v = fmaxf(v, __shfl_xor(v, 1, 16));
        v = fmaxf(v, __shfl_xor(v, 2, 16));
        v = fmaxf(v, __shfl_xor(v, 4, 16));
        v = fmaxf(v, __shfl_xor(v, 8, 16));
        float mn = fmaxf(m[r], v);
        alpha[r] = __builtin_amdgcn_exp2f(m[r] - mn);
        m[r] = mn;
      }
#pragma unroll
      for (int r = 0; r < 8; ++r) {
        S0[r] = __builtin_amdgcn_exp2f(S0[r] - m[r]);
        S1[r] = __builtin_amdgcn_exp2f(S1[r] - m[r]);
        float v = S0[r] + S1[r];
        v += __shfl_xor(v, 1, 16);
        v += __shfl_xor(v, 2, 16);
        v += __shfl_xor(v, 4, 16);
        v += __shfl_xor(v, 8, 16);
        l[r] = l[r] * alpha[r] + v;
      }
#pragma unroll
      for (int r = 0; r < 8; ++r) {
        O0[r] *= alpha[r]; O1[r] *= alpha[r];
        O2[r] *= alpha[r]; O3[r] *= alpha[r];
      }

      // ---- P (C layout f32) -> LDS -> A-fragment (bf16) ----
#pragma unroll
      for (int r = 0; r < 8; ++r) {
        int row = r + 8 * hi;
        pw[row * PT_STRIDE + nl]      = (bf16_t)S0[r];
        pw[row * PT_STRIDE + 16 + nl] = (bf16_t)S1[r];
      }
      asm volatile("s_wait_dscnt 0" ::: "memory"); // LDS ops in-order within wave
      v16bf pa = load_fragA(pw + nl * PT_STRIDE, hi);

      // ---- O += P @ V : V^T tile in LDS, column n = head dim ----
      O0 = wmma_bf16(pa, load_fragB(vtb + nl * VT_STRIDE, hi), O0);
      O1 = wmma_bf16(pa, load_fragB(vtb + (16 + nl) * VT_STRIDE, hi), O1);
      O2 = wmma_bf16(pa, load_fragB(vtb + (32 + nl) * VT_STRIDE, hi), O2);
      O3 = wmma_bf16(pa, load_fragB(vtb + (48 + nl) * VT_STRIDE, hi), O3);
    }

    __syncthreads();                               // all reads of buf done
    buf ^= 1;
  }

  // ---- epilogue: divide by row sums, store f32 ----
  float* outb = out + ((size_t)b * Tn + q0) * Hn;
#pragma unroll
  for (int r = 0; r < 8; ++r) {
    float inv = 1.0f / l[r];
    float* orow = outb + (size_t)(r + 8 * hi) * Hn + nl;
    orow[0]  = O0[r] * inv;
    orow[16] = O1[r] * inv;
    orow[32] = O2[r] * inv;
    orow[48] = O3[r] * inv;
  }
}

// ---------------------------------------------------------------------------
extern "C" void kernel_launch(void* const* d_in, const int* in_sizes, int n_in,
                              void* d_out, int out_size, void* d_ws, size_t ws_size,
                              hipStream_t stream) {
  (void)in_sizes; (void)n_in; (void)out_size; (void)ws_size;
  const float* x  = (const float*)d_in[0];
  const float* Wk = (const float*)d_in[1];
  const float* Wq = (const float*)d_in[2];
  const float* Wv = (const float*)d_in[3];
  float* out = (float*)d_out;

  char* ws = (char*)d_ws;
  // ws layout (bytes): Wt bf16 [3][64][1024] @0 (384 KiB), then K,Q bf16 [B,T,H]
  // (2 MiB each), then Vt bf16 [B,H,T] (2 MiB). Total ~6.4 MiB.
  bf16_t* Wt = (bf16_t*)(ws);
  bf16_t* Kb = (bf16_t*)(ws + 393216);
  bf16_t* Qb = (bf16_t*)(ws + 393216 + 2097152);
  bf16_t* Vt = (bf16_t*)(ws + 393216 + 2 * 2097152);

  wconv_kernel<<<768, 256, 0, stream>>>(Wk, Wq, Wv, Wt);
  proj_kernel <<<256, 256, 0, stream>>>(x, Wt, Kb, Qb, Vt);
  flash_kernel<<<256, 128, 0, stream>>>(Qb, Kb, Vt, out);
}
